// SpeciesIndexNet_57750130262456
// MI455X (gfx1250) — matrix operand order repaired and checked by reading, not compiled, not run on hardware
//
#include <hip/hip_runtime.h>
#include <cstdint>
#include <cstddef>

// ---------- types ----------
typedef __attribute__((ext_vector_type(16))) __bf16 v16bf;
typedef __attribute__((ext_vector_type(8)))  float  v8f;

#define SPECIES 4
#define PER_SP  50000
#define D_IN    256
#define D_H     512
#define D_OUT   256
#define TILE_M  128
#define PADH    8
#define THREADS 256
#define TILES_PER_SPECIES ((PER_SP + TILE_M - 1) / TILE_M)   // 391 (last tile 80 rows)

__device__ __forceinline__ unsigned short f32_to_bf16_rne(float f) {
  unsigned int u = __float_as_uint(f);
  u += 0x7FFFu + ((u >> 16) & 1u);          // round to nearest even
  return (unsigned short)(u >> 16);
}

// ---------------------------------------------------------------------------
// Prep: f32 weights [S][K][N] -> bf16, pre-swizzled into WMMA B-operand blocks.
// Block = (nt, kt) covering K32 x N16; within block, element (k,n) lives at
//   lane = (n&15) + 16*((k>>4)&1),  half = k&15,  offset = lane*16 + half
// so each wave lane later loads its 16 halves (32 B) contiguously.
// ---------------------------------------------------------------------------
__global__ void pack_weights_bf16(const float* __restrict__ W,
                                  unsigned short* __restrict__ Wp,
                                  int K, int N, int total) {
  int i = blockIdx.x * blockDim.x + threadIdx.x;
  if (i >= total) return;
  int kn = K * N;
  int s  = i / kn;
  int r  = i - s * kn;
  int k  = r / N;
  int n  = r - k * N;
  int kt   = k >> 5;
  int nt   = n >> 4;
  int lane = (n & 15) + (((k >> 4) & 1) << 4);
  int h    = k & 15;
  size_t off = (size_t)s * kn + ((size_t)(nt * (K >> 5) + kt)) * 512 + (size_t)lane * 16 + h;
  Wp[off] = f32_to_bf16_rne(W[i]);
}

// ---------------------------------------------------------------------------
// GEMM accumulate over 128 rows: acc[2][8] (+)= A(128 x 32*KSTEPS, LDS bf16)
//                                             x B(packed global bf16 blocks)
// This pass owns N-tiles {ntBase, ntBase+8}.  Per K-step: B (2 blocks, 16
// VGPRs) hoisted, A streamed in two 4-M-tile halves (32 VGPRs live) -> each
// 1KB weight block feeds 8 WMMAs (128 flops per L2 byte).
// ---------------------------------------------------------------------------
template<int KSTEPS>
__device__ __forceinline__ void gemm_acc8(
    const unsigned short* __restrict__ inL, int sin,
    const unsigned short* __restrict__ Wp,
    int ntBase, int lane, v8f (&acc)[2][8]) {
  const int m  = lane & 15;
  const int kh = lane >> 4;
  for (int kt = 0; kt < KSTEPS; ++kt) {
    union { uint4 q[2]; v16bf v; } ub[2];
#pragma unroll
    for (int j = 0; j < 2; ++j) {
      const unsigned short* bp =
          Wp + ((size_t)((ntBase + 8 * j) * KSTEPS + kt)) * 512 + (size_t)lane * 16;
      ub[j].q[0] = *(const uint4*)(bp);
      ub[j].q[1] = *(const uint4*)(bp + 8);
    }
#pragma unroll
    for (int mh = 0; mh < 2; ++mh) {
      v16bf A[4];
#pragma unroll
      for (int mtl = 0; mtl < 4; ++mtl) {
        const unsigned short* p =
            inL + (size_t)((mh * 4 + mtl) * 16 + m) * sin + kt * 32 + kh * 16;
        union { uint4 q[2]; v16bf v; } ua;
        ua.q[0] = *(const uint4*)(p);          // K = kb .. kb+7
        ua.q[1] = *(const uint4*)(p + 8);      // K = kb+8 .. kb+15
        A[mtl] = ua.v;
      }
#pragma unroll
      for (int j = 0; j < 2; ++j)
#pragma unroll
        for (int mtl = 0; mtl < 4; ++mtl)
          acc[j][mh * 4 + mtl] = __builtin_amdgcn_wmma_f32_16x16x32_bf16(
              false, A[mtl], false, ub[j].v, (short)0, acc[j][mh * 4 + mtl],
              false, false);
    }
  }
}

// bias + SiLU + f32->bf16 -> LDS (row-major, padded)
__device__ __forceinline__ void epilogue_silu(
    v8f (&acc)[2][8], const float* __restrict__ bias, int ntBase,
    unsigned short* __restrict__ outL, int sout, int lane) {
  const int m  = lane & 15;
  const int kh = lane >> 4;
#pragma unroll
  for (int j = 0; j < 2; ++j) {
    int col = (ntBase + 8 * j) * 16 + m;
    float bv = bias[col];
#pragma unroll
    for (int mt = 0; mt < 8; ++mt)
#pragma unroll
      for (int r = 0; r < 8; ++r) {
        float v = acc[j][mt][r] + bv;
        v = v / (1.0f + __expf(-v));           // SiLU
        outL[(size_t)(mt * 16 + 8 * kh + r) * sout + col] = f32_to_bf16_rne(v);
      }
  }
}

// ---------------------------------------------------------------------------
// Fused 3-layer species MLP. One workgroup = 128-atom tile of one species.
// LDS (266.8 KB of 320 KB/WGP), overlapped:
//   region0: x (bf16 128x264) ... later h2 (bf16 128x520)
//   region1: h1 (bf16 128x520)
// ---------------------------------------------------------------------------
__global__ __launch_bounds__(THREADS)
void species_mlp_fused(const float* __restrict__ emb,
                       const int*   __restrict__ idx,
                       const unsigned short* __restrict__ W1p,
                       const float* __restrict__ b1,
                       const unsigned short* __restrict__ W2p,
                       const float* __restrict__ b2,
                       const unsigned short* __restrict__ W3p,
                       const float* __restrict__ b3,
                       float* __restrict__ out) {
  extern __shared__ __align__(16) unsigned short smem[];
  const int HREG = TILE_M * (D_H + PADH);        // 66560 halves = 133120 B
  unsigned short* bufX  = smem;                  // region0 (first 128*264 halves)
  unsigned short* bufH1 = smem + HREG;           // region1
  unsigned short* bufH2 = smem;                  // region0 (overwrites x, post-sync)
  int*            rowid = (int*)(smem + 2 * HREG);

  const int s    = blockIdx.x / TILES_PER_SPECIES;
  const int tile = blockIdx.x % TILES_PER_SPECIES;
  const int t    = threadIdx.x;
  const int lane = t & 31;
  const int wid  = t >> 5;
  int validRows = PER_SP - tile * TILE_M;
  if (validRows > TILE_M) validRows = TILE_M;

  if (t < TILE_M) {
    int p = tile * TILE_M + t;
    if (p >= PER_SP) p = PER_SP - 1;             // clamp (rows discarded later)
    rowid[t] = idx[s * PER_SP + p];
  }
  __syncthreads();

  // ---- gather atoms + f32 -> bf16 into LDS ----
  const int SX = D_IN + PADH;
#pragma unroll 4
  for (int i = t; i < TILE_M * D_IN; i += THREADS) {
    int r = i >> 8;                              // D_IN == 256
    int c = i & (D_IN - 1);
    float v = emb[(size_t)rowid[r] * D_IN + c];
    bufX[r * SX + c] = f32_to_bf16_rne(v);
  }
  __syncthreads();

  const int SH = D_H + PADH;
  const int m  = lane & 15;

  // ---- layer 1: [128,256] x [256,512] + b1, SiLU  (two N-passes) ----
#pragma unroll 1
  for (int pass = 0; pass < 2; ++pass) {
    v8f acc[2][8];
#pragma unroll
    for (int j = 0; j < 2; ++j)
#pragma unroll
      for (int mt = 0; mt < 8; ++mt) acc[j][mt] = (v8f){};
    int ntBase = wid + 16 * pass;
    gemm_acc8<D_IN / 32>(bufX, SX, W1p + (size_t)s * D_IN * D_H, ntBase, lane, acc);
    epilogue_silu(acc, b1 + s * D_H, ntBase, bufH1, SH, lane);
  }
  __syncthreads();

  // ---- layer 2: [128,512] x [512,512] + b2, SiLU  (two N-passes) ----
#pragma unroll 1
  for (int pass = 0; pass < 2; ++pass) {
    v8f acc[2][8];
#pragma unroll
    for (int j = 0; j < 2; ++j)
#pragma unroll
      for (int mt = 0; mt < 8; ++mt) acc[j][mt] = (v8f){};
    int ntBase = wid + 16 * pass;
    gemm_acc8<D_H / 32>(bufH1, SH, W2p + (size_t)s * D_H * D_H, ntBase, lane, acc);
    epilogue_silu(acc, b2 + s * D_H, ntBase, bufH2, SH, lane);
  }
  __syncthreads();

  // ---- layer 3: [128,512] x [512,256] + b3, scatter f32 to global ----
  {
    v8f acc[2][8];
#pragma unroll
    for (int j = 0; j < 2; ++j)
#pragma unroll
      for (int mt = 0; mt < 8; ++mt) acc[j][mt] = (v8f){};
    gemm_acc8<D_H / 32>(bufH2, SH, W3p + (size_t)s * D_H * D_OUT, wid, lane, acc);
    const float* bb = b3 + s * D_OUT;
    const int kh = lane >> 4;
#pragma unroll
    for (int j = 0; j < 2; ++j) {
      int col = (wid + 8 * j) * 16 + m;
      float bv = bb[col];
#pragma unroll
      for (int mt = 0; mt < 8; ++mt)
#pragma unroll
        for (int r = 0; r < 8; ++r) {
          int row = mt * 16 + 8 * kh + r;
          if (row < validRows)
            out[(size_t)rowid[row] * D_OUT + col] = acc[j][mt][r] + bv;
        }
    }
  }
}

// ---------------------------------------------------------------------------
extern "C" void kernel_launch(void* const* d_in, const int* in_sizes, int n_in,
                              void* d_out, int out_size, void* d_ws, size_t ws_size,
                              hipStream_t stream) {
  (void)in_sizes; (void)n_in; (void)out_size; (void)ws_size;
  // setup_inputs order: species, embedding, idx, W1, b1, W2, b2, W3, b3
  const float* emb = (const float*)d_in[1];
  const int*   idx = (const int*)d_in[2];
  const float* W1  = (const float*)d_in[3];
  const float* b1  = (const float*)d_in[4];
  const float* W2  = (const float*)d_in[5];
  const float* b2  = (const float*)d_in[6];
  const float* W3  = (const float*)d_in[7];
  const float* b3  = (const float*)d_in[8];
  float* out = (float*)d_out;

  // workspace: bf16-packed, B-operand-swizzled weights (4 MB total, L2-resident)
  unsigned short* W1p = (unsigned short*)d_ws;
  unsigned short* W2p = W1p + (size_t)SPECIES * D_IN * D_H;
  unsigned short* W3p = W2p + (size_t)SPECIES * D_H * D_H;

  int t1 = SPECIES * D_IN * D_H;
  pack_weights_bf16<<<(t1 + 255) / 256, 256, 0, stream>>>(W1, W1p, D_IN, D_H, t1);
  int t2 = SPECIES * D_H * D_H;
  pack_weights_bf16<<<(t2 + 255) / 256, 256, 0, stream>>>(W2, W2p, D_H, D_H, t2);
  int t3 = SPECIES * D_H * D_OUT;
  pack_weights_bf16<<<(t3 + 255) / 256, 256, 0, stream>>>(W3, W3p, D_H, D_OUT, t3);

  size_t shmem = 2 * (size_t)TILE_M * (D_H + PADH) * sizeof(unsigned short)
               + TILE_M * sizeof(int);   // 266,752 B of the 320 KB/WGP LDS
  hipFuncSetAttribute(reinterpret_cast<const void*>(species_mlp_fused),
                      hipFuncAttributeMaxDynamicSharedMemorySize, (int)shmem);

  species_mlp_fused<<<SPECIES * TILES_PER_SPECIES, THREADS, shmem, stream>>>(
      emb, idx, W1p, b1, W2p, b2, W3p, b3, out);
}
